// ConViTCast_20315195310358
// MI455X (gfx1250) — compile-verified, matrix operand order checked by reading, not run on hardware
//
#include <hip/hip_runtime.h>
#include <hip/hip_bf16.h>
#include <math.h>

// ---------------------------------------------------------------------------
// ClimODE-style ViT forward for MI455X (gfx1250, wave32, WMMA).
// All dense matmuls go through v_wmma_f32_16x16x32_f16 (f16 in, f32 acc).
// Hot ODE weights are pre-converted to f16 once per depth (reused 40x) so the
// inner loop loads them with plain b128 loads instead of re-converting.
// ---------------------------------------------------------------------------

typedef __attribute__((ext_vector_type(16))) _Float16 v16h;
typedef __attribute__((ext_vector_type(8)))  _Float16 v8h;
typedef __attribute__((ext_vector_type(8)))  float    v8f;

#define CB   2
#define CV   5
#define CH   32
#define CW   64
#define CP   2
#define CD   1024
#define CNH  16
#define CHD  64
#define CL   512            // (H/P)*(W/P)
#define CM   (CB * CL)      // 1024 token rows
#define C3D  (3 * CD)
#define CRES 0.01f
#define CEPS 1e-5f

// ------------------------------ WMMA helpers -------------------------------
// A-frag (16x32 f16, M x K) / B-frag (32x16 f16, K x N) share the same
// per-lane gather when both operands are row-major with K contiguous
// (B element (k,n) = W[n][k] for C = A * W^T).
// ISA 7.12.2 wave32 layout: lane = (idx & 15) + 16*kgroup;
//   halves j=0..7  -> k = kgroup*8 + j
//   halves j=8..15 -> k = 16 + kgroup*8 + j
__device__ __forceinline__ v16h load_frag_f32(const float* __restrict__ base,
                                              int ld, int row, int k0, int lane) {
    const int kg = (lane >> 4) << 3;  // 0 or 8
    const float* p = base + (size_t)(row + (lane & 15)) * ld + k0 + kg;
    v16h f;
#pragma unroll
    for (int j = 0; j < 8; ++j) f[j] = (_Float16)p[j];
#pragma unroll
    for (int j = 0; j < 8; ++j) f[j + 8] = (_Float16)p[16 + j];
    return f;
}

// f16-resident operand: two 16-byte vector loads, no conversion VALU.
__device__ __forceinline__ v16h load_frag_f16(const _Float16* __restrict__ base,
                                              int ld, int row, int k0, int lane) {
    const int kg = (lane >> 4) << 3;
    const _Float16* p = base + (size_t)(row + (lane & 15)) * ld + k0 + kg;
    const v8h lo = *(const v8h*)(p);
    const v8h hi = *(const v8h*)(p + 16);
    v16h f;
#pragma unroll
    for (int j = 0; j < 8; ++j) { f[j] = lo[j]; f[j + 8] = hi[j]; }
    return f;
}

__device__ __forceinline__ v8f wmma_f16(v16h a, v16h b, v8f c) {
    return __builtin_amdgcn_wmma_f32_16x16x32_f16(false, a, false, b,
                                                  (short)0, c, false, false);
}

__device__ __forceinline__ float gelu_exact(float x) {
    return 0.5f * x * (1.0f + erff(x * 0.70710678118654752f));
}

__device__ __forceinline__ float apply_act(float v, int ACT) {
    if (ACT == 1) return fmaxf(v, 0.0f);
    if (ACT == 2) return gelu_exact(v);
    return v;
}

// ----------------------- GEMM: C[M,N] = A[M,K] @ W[N,K]^T + bias -----------
// 256 threads = 8 waves; macro tile 128(M) x 128(N); wave tile 64(M) x 32(N)
// = 4x2 WMMA tiles, accumulators live in VGPRs for the whole K loop.
template <int ACT>  // 0 = none, 1 = relu, 2 = exact gelu
__global__ __launch_bounds__(256)
void gemm_bias_act(const float* __restrict__ A, const float* __restrict__ W,
                   const float* __restrict__ bias, float* __restrict__ C,
                   int M, int N, int K) {
    const int lane = threadIdx.x & 31;
    const int wave = threadIdx.x >> 5;          // 0..7
    const int wm   = wave & 1;                  // 2 waves along M
    const int wn   = wave >> 1;                 // 4 waves along N
    const int bm   = blockIdx.y * 128 + wm * 64;
    const int bn   = blockIdx.x * 128 + wn * 32;

    v8f zero = {};
    v8f acc[4][2];
#pragma unroll
    for (int mt = 0; mt < 4; ++mt)
#pragma unroll
        for (int nt = 0; nt < 2; ++nt) acc[mt][nt] = zero;

    for (int k0 = 0; k0 < K; k0 += 32) {
        v16h af[4], bf[2];
#pragma unroll
        for (int mt = 0; mt < 4; ++mt)
            af[mt] = load_frag_f32(A, K, bm + mt * 16, k0, lane);
#pragma unroll
        for (int nt = 0; nt < 2; ++nt)
            bf[nt] = load_frag_f32(W, K, bn + nt * 16, k0, lane);
        if (k0 + 32 < K) {  // pull next weight tile toward L2/L0 (global_prefetch_b8)
            __builtin_prefetch(W + (size_t)(bn + (lane & 15)) * K + k0 + 32, 0, 1);
        }
#pragma unroll
        for (int mt = 0; mt < 4; ++mt)
#pragma unroll
            for (int nt = 0; nt < 2; ++nt)
                acc[mt][nt] = wmma_f16(af[mt], bf[nt], acc[mt][nt]);
    }

    // Epilogue. C/D layout: lane = n(0..15) + 16*(m>=8); VGPR r -> m = r + 8*hi.
    const int nlo = lane & 15;
    const int mhi = (lane >> 4) << 3;
#pragma unroll
    for (int nt = 0; nt < 2; ++nt) {
        const int n  = bn + nt * 16 + nlo;
        const float bv = bias ? bias[n] : 0.0f;
#pragma unroll
        for (int mt = 0; mt < 4; ++mt) {
            const int mrow = bm + mt * 16 + mhi;
#pragma unroll
            for (int r = 0; r < 8; ++r)
                C[(size_t)(mrow + r) * N + n] = apply_act(acc[mt][nt][r] + bv, ACT);
        }
    }
}

// Same GEMM but with pre-converted f16 weights (ODE hot path; W reused 40x).
template <int ACT>
__global__ __launch_bounds__(256)
void gemm_bias_act_wf16(const float* __restrict__ A, const _Float16* __restrict__ W,
                        const float* __restrict__ bias, float* __restrict__ C,
                        int M, int N, int K) {
    const int lane = threadIdx.x & 31;
    const int wave = threadIdx.x >> 5;
    const int wm   = wave & 1;
    const int wn   = wave >> 1;
    const int bm   = blockIdx.y * 128 + wm * 64;
    const int bn   = blockIdx.x * 128 + wn * 32;

    v8f zero = {};
    v8f acc[4][2];
#pragma unroll
    for (int mt = 0; mt < 4; ++mt)
#pragma unroll
        for (int nt = 0; nt < 2; ++nt) acc[mt][nt] = zero;

    for (int k0 = 0; k0 < K; k0 += 32) {
        v16h af[4], bf[2];
#pragma unroll
        for (int mt = 0; mt < 4; ++mt)
            af[mt] = load_frag_f32(A, K, bm + mt * 16, k0, lane);
#pragma unroll
        for (int nt = 0; nt < 2; ++nt)
            bf[nt] = load_frag_f16(W, K, bn + nt * 16, k0, lane);
        if (k0 + 32 < K) {
            __builtin_prefetch(W + (size_t)(bn + (lane & 15)) * K + k0 + 32, 0, 1);
        }
#pragma unroll
        for (int mt = 0; mt < 4; ++mt)
#pragma unroll
            for (int nt = 0; nt < 2; ++nt)
                acc[mt][nt] = wmma_f16(af[mt], bf[nt], acc[mt][nt]);
    }

    const int nlo = lane & 15;
    const int mhi = (lane >> 4) << 3;
#pragma unroll
    for (int nt = 0; nt < 2; ++nt) {
        const int n  = bn + nt * 16 + nlo;
        const float bv = bias ? bias[n] : 0.0f;
#pragma unroll
        for (int mt = 0; mt < 4; ++mt) {
            const int mrow = bm + mt * 16 + mhi;
#pragma unroll
            for (int r = 0; r < 8; ++r)
                C[(size_t)(mrow + r) * N + n] = apply_act(acc[mt][nt][r] + bv, ACT);
        }
    }
}

// f32 -> f16 bulk conversion (for per-depth ODE weight staging)
__global__ __launch_bounds__(256)
void f32_to_f16_kernel(const float* __restrict__ in, _Float16* __restrict__ out, int n) {
    const int i = blockIdx.x * 256 + threadIdx.x;
    if (i < n) out[i] = (_Float16)in[i];
}

// -------------------- patch embed (fused into xb layout) -------------------
__global__ __launch_bounds__(256)
void patch_embed_kernel(const float* __restrict__ x,
                        const float* __restrict__ pw,
                        const float* __restrict__ pb,
                        const float* __restrict__ ve,
                        float* __restrict__ xb) {
    const int idx = blockIdx.x * 256 + threadIdx.x;
    if (idx >= CM * CV * CD) return;
    const int d = idx & (CD - 1);
    const int t = idx >> 10;
    const int v = t % CV;
    const int m = t / CV;
    const int b  = m >> 9;        // / L
    const int l  = m & (CL - 1);
    const int hi = l >> 5;        // / (W/P)
    const int wi = l & 31;
    const float* xp = x + ((size_t)(b * CV + v) * CH + 2 * hi) * CW + 2 * wi;
    const float* wp = pw + ((size_t)v * CD + d) * 4;
    float s = xp[0] * wp[0] + xp[1] * wp[1] + xp[CW] * wp[2] + xp[CW + 1] * wp[3];
    s += pb[v * CD + d] + ve[v * CD + d];
    xb[idx] = s;
}

// -------- q = var_query @ Wq^T + bq  (single row; plain dot products) ------
__global__ __launch_bounds__(256)
void qvec_kernel(const float* __restrict__ vq, const float* __restrict__ wq,
                 const float* __restrict__ bq, float* __restrict__ q) {
    const int n = blockIdx.x * 256 + threadIdx.x;
    if (n >= CD) return;
    const float* wr = wq + (size_t)n * CD;
    float s = 0.f;
    for (int k = 0; k < CD; ++k) s += vq[k] * wr[k];
    q[n] = s + bq[n];
}

// ---------------- variable-aggregation attention (V = 5 tokens) ------------
__global__ __launch_bounds__(256)
void agg_attn_kernel(const float* __restrict__ q, const float* __restrict__ kv,
                     float* __restrict__ agg) {
    const int t = blockIdx.x * 256 + threadIdx.x;
    if (t >= CM * CNH) return;
    const int h = t & (CNH - 1);
    const int m = t >> 4;
    const float* qh = q + h * CHD;
    float sc[CV], mx = -3.0e38f;
    for (int v = 0; v < CV; ++v) {
        const float* kp = kv + ((size_t)(m * CV + v) * 2048) + h * CHD;
        float s = 0.f;
        for (int d = 0; d < CHD; ++d) s += qh[d] * kp[d];
        s *= 0.125f;  // 1/sqrt(64)
        sc[v] = s;
        mx = fmaxf(mx, s);
    }
    float sum = 0.f;
    for (int v = 0; v < CV; ++v) { sc[v] = __expf(sc[v] - mx); sum += sc[v]; }
    const float inv = 1.0f / sum;
    float* op = agg + (size_t)m * CD + h * CHD;
    for (int d = 0; d < CHD; ++d) {
        float o = 0.f;
        for (int v = 0; v < CV; ++v)
            o += sc[v] * kv[((size_t)(m * CV + v) * 2048) + 1024 + h * CHD + d];
        op[d] = o * inv;
    }
}

// ------------- xs += pos_embed[l] + lead_times[b]*lt_w + lt_b --------------
__global__ __launch_bounds__(256)
void pos_lt_kernel(float* __restrict__ xs, const float* __restrict__ pos,
                   const float* __restrict__ ltw, const float* __restrict__ ltb,
                   const float* __restrict__ lt) {
    const int i = blockIdx.x * 256 + threadIdx.x;
    if (i >= CM * CD) return;
    const int d  = i & (CD - 1);
    const int ml = i >> 10;
    const int l  = ml & (CL - 1);
    const int b  = ml >> 9;
    xs[i] += pos[(size_t)l * CD + d] + lt[b] * ltw[d] + ltb[d];
}

// ----------------------- patch self-attention (WMMA) -----------------------
// One wave per (b, head, 16-query tile). S = Q*K^T via WMMA into LDS,
// row softmax, then O = P*V via WMMA.
__global__ __launch_bounds__(32)
void attn_kernel(const float* __restrict__ qkv, float* __restrict__ o) {
    __shared__ float s_s[16 * CL];  // 32 KB scores
    const int lane = threadIdx.x & 31;
    const int qt = blockIdx.x & 31;
    const int h  = (blockIdx.x >> 5) & 15;
    const int b  = blockIdx.x >> 9;
    const int l0 = qt * 16;
    const float* base = qkv + (size_t)b * CL * C3D + h * CHD;  // q; k:+1024; v:+2048

    // Q fragments (scaled by 1/sqrt(hd) during f32->f16 conversion)
    v16h aq[2];
    {
        const int kg = (lane >> 4) << 3;
        const float* qrow = base + (size_t)(l0 + (lane & 15)) * C3D;
#pragma unroll
        for (int c = 0; c < 2; ++c) {
#pragma unroll
            for (int j = 0; j < 8; ++j)
                aq[c][j] = (_Float16)(qrow[c * 32 + kg + j] * 0.125f);
#pragma unroll
            for (int j = 0; j < 8; ++j)
                aq[c][j + 8] = (_Float16)(qrow[c * 32 + 16 + kg + j] * 0.125f);
        }
    }
    // S tiles
    v8f zero = {};
    for (int kt = 0; kt < CL / 16; ++kt) {
        const int kg = (lane >> 4) << 3;
        const float* krow = base + 1024 + (size_t)(kt * 16 + (lane & 15)) * C3D;
        v16h bk[2];
#pragma unroll
        for (int c = 0; c < 2; ++c) {
#pragma unroll
            for (int j = 0; j < 8; ++j) bk[c][j] = (_Float16)krow[c * 32 + kg + j];
#pragma unroll
            for (int j = 0; j < 8; ++j) bk[c][j + 8] = (_Float16)krow[c * 32 + 16 + kg + j];
        }
        v8f acc = zero;
        acc = wmma_f16(aq[0], bk[0], acc);
        acc = wmma_f16(aq[1], bk[1], acc);
        const int col = kt * 16 + (lane & 15);
        const int mb  = (lane >> 4) << 3;
#pragma unroll
        for (int r = 0; r < 8; ++r) s_s[(mb + r) * CL + col] = acc[r];
    }
    __syncthreads();
    // softmax per query row (lanes 0..15)
    if (lane < 16) {
        float* row = s_s + lane * CL;
        float mx = -3.0e38f;
        for (int c = 0; c < CL; ++c) mx = fmaxf(mx, row[c]);
        float sum = 0.f;
        for (int c = 0; c < CL; ++c) { float e = __expf(row[c] - mx); row[c] = e; sum += e; }
        const float inv = 1.0f / sum;
        for (int c = 0; c < CL; ++c) row[c] *= inv;
    }
    __syncthreads();
    // O = P @ V
    const float* vbase = base + 2048;
    for (int nt = 0; nt < 4; ++nt) {
        v8f acc = zero;
        for (int kc = 0; kc < CL / 32; ++kc) {
            const int kg = (lane >> 4) << 3;
            const float* prow = s_s + (lane & 15) * CL + kc * 32;
            v16h ap, bv;
#pragma unroll
            for (int j = 0; j < 8; ++j) ap[j] = (_Float16)prow[kg + j];
#pragma unroll
            for (int j = 0; j < 8; ++j) ap[j + 8] = (_Float16)prow[16 + kg + j];
            const int d = nt * 16 + (lane & 15);
#pragma unroll
            for (int j = 0; j < 8; ++j)
                bv[j] = (_Float16)vbase[(size_t)(kc * 32 + kg + j) * C3D + d];
#pragma unroll
            for (int j = 0; j < 8; ++j)
                bv[j + 8] = (_Float16)vbase[(size_t)(kc * 32 + 16 + kg + j) * C3D + d];
            acc = wmma_f16(ap, bv, acc);
        }
        const int n  = nt * 16 + (lane & 15);
        const int mb = (lane >> 4) << 3;
#pragma unroll
        for (int r = 0; r < 8; ++r)
            o[(size_t)(b * CL + l0 + mb + r) * CD + h * CHD + n] = acc[r];
    }
}

// ----------------------- LayerNorm: out = LN(in [+ res]) -------------------
__global__ __launch_bounds__(256)
void ln_kernel(float* __restrict__ out, const float* __restrict__ in,
               const float* __restrict__ res, const float* __restrict__ g,
               const float* __restrict__ b) {
    __shared__ float red[256];
    const int row = blockIdx.x;
    const int tid = threadIdx.x;
    const float* ip = in + (size_t)row * CD;
    const float* rp = res ? res + (size_t)row * CD : nullptr;
    float v[4];
    float s = 0.f;
#pragma unroll
    for (int e = 0; e < 4; ++e) {
        const int d = tid + e * 256;
        float x = ip[d];
        if (rp) x += rp[d];
        v[e] = x;
        s += x;
    }
    red[tid] = s; __syncthreads();
    for (int st = 128; st > 0; st >>= 1) { if (tid < st) red[tid] += red[tid + st]; __syncthreads(); }
    const float mean = red[0] * (1.0f / CD);
    __syncthreads();
    float s2 = 0.f;
#pragma unroll
    for (int e = 0; e < 4; ++e) { float d0 = v[e] - mean; s2 += d0 * d0; }
    red[tid] = s2; __syncthreads();
    for (int st = 128; st > 0; st >>= 1) { if (tid < st) red[tid] += red[tid + st]; __syncthreads(); }
    const float inv = rsqrtf(red[0] * (1.0f / CD) + CEPS);
#pragma unroll
    for (int e = 0; e < 4; ++e) {
        const int d = tid + e * 256;
        out[(size_t)row * CD + d] = (v[e] - mean) * inv * g[d] + b[d];
    }
}

// out = y + a * k
__global__ __launch_bounds__(256)
void axpy_kernel(float* __restrict__ out, const float* __restrict__ y,
                 const float* __restrict__ k, float a, int n) {
    const int i = blockIdx.x * 256 + threadIdx.x;
    if (i < n) out[i] = y[i] + a * k[i];
}

// RK4 combine (in-place on y) + per-batch trajectory readout
__global__ __launch_bounds__(256)
void rk4_kernel(float* __restrict__ y, const float* __restrict__ k1,
                const float* __restrict__ k2, const float* __restrict__ k3,
                const float* __restrict__ k4, float* __restrict__ ode_out,
                const float* __restrict__ lt, int s) {
    const int i = blockIdx.x * 256 + threadIdx.x;
    if (i >= CM * CD) return;
    const float y2 = y[i] + (CRES / 6.0f) * (k1[i] + 2.f * k2[i] + 2.f * k3[i] + k4[i]);
    y[i] = y2;
    const int b = i >> 19;  // / (L*D)
    const int idx = (int)rintf(lt[b] * 100.0f);  // lead / RES
    if (idx == s + 1) ode_out[i] = y2;
}

// -------------------- head projection (N = V*P*P = 20) + unpatchify --------
__global__ __launch_bounds__(256)
void head_kernel(const float* __restrict__ a, const float* __restrict__ w,
                 const float* __restrict__ bias, float* __restrict__ out) {
    const int t = blockIdx.x * 256 + threadIdx.x;
    if (t >= CM * 20) return;
    const int n = t % 20;
    const int m = t / 20;
    const float* ar = a + (size_t)m * CD;
    const float* wr = w + (size_t)n * CD;
    float s = 0.f;
    for (int k = 0; k < CD; ++k) s += ar[k] * wr[k];
    s += bias[n];
    const int p = n / 10, q = (n % 10) / 5, v = n % 5;
    const int b = m >> 9, l = m & (CL - 1), hi = l >> 5, wi = l & 31;
    out[((size_t)(b * CV + v) * CH + 2 * hi + p) * CW + 2 * wi + q] = s;
}

// ------------------------------- host side ---------------------------------
static void launch_gemm(const float* A, const float* W, const float* bias,
                        float* C, int M, int N, int K, int act, hipStream_t s) {
    dim3 grid(N / 128, M / 128), blk(256);
    switch (act) {
        case 0: gemm_bias_act<0><<<grid, blk, 0, s>>>(A, W, bias, C, M, N, K); break;
        case 1: gemm_bias_act<1><<<grid, blk, 0, s>>>(A, W, bias, C, M, N, K); break;
        default: gemm_bias_act<2><<<grid, blk, 0, s>>>(A, W, bias, C, M, N, K); break;
    }
}

static void launch_gemm_wf16(const float* A, const _Float16* W, const float* bias,
                             float* C, int M, int N, int K, int act, hipStream_t s) {
    dim3 grid(N / 128, M / 128), blk(256);
    switch (act) {
        case 0: gemm_bias_act_wf16<0><<<grid, blk, 0, s>>>(A, W, bias, C, M, N, K); break;
        case 1: gemm_bias_act_wf16<1><<<grid, blk, 0, s>>>(A, W, bias, C, M, N, K); break;
        default: gemm_bias_act_wf16<2><<<grid, blk, 0, s>>>(A, W, bias, C, M, N, K); break;
    }
}

extern "C" void kernel_launch(void* const* d_in, const int* in_sizes, int n_in,
                              void* d_out, int out_size, void* d_ws, size_t ws_size,
                              hipStream_t stream) {
    (void)in_sizes; (void)n_in; (void)out_size; (void)ws_size;
    const float* x         = (const float*)d_in[0];
    const float* lead      = (const float*)d_in[1];
    const float* patch_w   = (const float*)d_in[2];
    const float* patch_b   = (const float*)d_in[3];
    const float* var_embed = (const float*)d_in[4];
    const float* var_query = (const float*)d_in[5];
    const float* agg_in_w  = (const float*)d_in[6];
    const float* agg_in_b  = (const float*)d_in[7];
    const float* agg_out_w = (const float*)d_in[8];
    const float* agg_out_b = (const float*)d_in[9];
    const float* pos_embed = (const float*)d_in[10];
    const float* lt_w      = (const float*)d_in[11];
    const float* lt_b      = (const float*)d_in[12];
    const float* blk_qkv_w = (const float*)d_in[13];
    const float* blk_qkv_b = (const float*)d_in[14];
    const float* blk_out_w = (const float*)d_in[15];
    const float* blk_out_b = (const float*)d_in[16];
    const float* n1_g      = (const float*)d_in[17];
    const float* n1_b      = (const float*)d_in[18];
    const float* n2_g      = (const float*)d_in[19];
    const float* n2_b      = (const float*)d_in[20];
    const float* ode_w1    = (const float*)d_in[21];
    const float* ode_b1    = (const float*)d_in[22];
    const float* ode_w2    = (const float*)d_in[23];
    const float* ode_b2    = (const float*)d_in[24];
    const float* ode_ng    = (const float*)d_in[25];
    const float* ode_nb    = (const float*)d_in[26];
    const float* norm_g    = (const float*)d_in[27];
    const float* norm_b    = (const float*)d_in[28];
    const float* h1_w      = (const float*)d_in[29];
    const float* h1_b      = (const float*)d_in[30];
    const float* h2_w      = (const float*)d_in[31];
    const float* h2_b      = (const float*)d_in[32];
    const float* hf_w      = (const float*)d_in[33];
    const float* hf_b      = (const float*)d_in[34];
    float* outp = (float*)d_out;
    float* f = (float*)d_ws;

    // workspace layout (floats); peak ~71 MB, phase-reused
    float* xb   = f + 0;          // [5120,1024]  phase 1
    float* kvb  = f + 5242880;    // [5120,2048]
    float* agg  = f + 15728640;   // [1024,1024]
    float* xs   = f + 16777216;   // [1024,1024]  persists
    float* qv   = f + 17825792;   // [1024]
    float* qkv  = f + 0;          // [1024,3072]  phase 2 (reuses phase-1 region)
    float* obuf = f + 3145728;
    float* t1   = f + 4194304;
    float* t2   = f + 5242880;
    float* k1   = f + 6291456;
    float* k2   = f + 7340032;
    float* k3   = f + 8388608;
    float* k4   = f + 9437184;
    float* ytmp = f + 10485760;
    float* ybuf = f + 11534336;
    float* oout = f + 12582912;
    _Float16* wh1 = (_Float16*)(f + 13631488);  // 1M halves (f16 ODE w1)
    _Float16* wh2 = (_Float16*)(f + 14155776);  // 1M halves (f16 ODE w2)
    float* hb1  = f + 0;          // phase 3
    float* hb2  = f + 1048576;

    const size_t TOK_BYTES = (size_t)CM * CD * sizeof(float);
    const int NTOK = CM * CD;              // 1,048,576
    const int EWG  = (NTOK + 255) / 256;

    // ---- phase 1: tokenize + aggregate variables ----
    patch_embed_kernel<<<(CM * CV * CD + 255) / 256, 256, 0, stream>>>(
        x, patch_w, patch_b, var_embed, xb);
    qvec_kernel<<<4, 256, 0, stream>>>(var_query, agg_in_w, agg_in_b, qv);
    // K,V projections in one WMMA GEMM: rows D..3D of agg_in_w
    launch_gemm(xb, agg_in_w + (size_t)CD * CD, agg_in_b + CD, kvb,
                CM * CV, 2 * CD, CD, 0, stream);
    agg_attn_kernel<<<(CM * CNH + 255) / 256, 256, 0, stream>>>(qv, kvb, agg);
    launch_gemm(agg, agg_out_w, agg_out_b, xs, CM, CD, CD, 0, stream);
    pos_lt_kernel<<<EWG, 256, 0, stream>>>(xs, pos_embed, lt_w, lt_b, lead);

    // ---- phase 2: DEPTH blocks (attention + RK4 ODE) ----
    for (int i = 0; i < 2; ++i) {
        launch_gemm(xs, blk_qkv_w + (size_t)i * C3D * CD, blk_qkv_b + (size_t)i * C3D,
                    qkv, CM, C3D, CD, 0, stream);
        attn_kernel<<<CB * CNH * (CL / 16), 32, 0, stream>>>(qkv, obuf);
        launch_gemm(obuf, blk_out_w + (size_t)i * CD * CD, blk_out_b + (size_t)i * CD,
                    t2, CM, CD, CD, 0, stream);
        ln_kernel<<<CM, 256, 0, stream>>>(xs, t2, xs, n1_g + (size_t)i * CD, n1_b + (size_t)i * CD);

        const float* b1 = ode_b1 + (size_t)i * CD;
        const float* b2 = ode_b2 + (size_t)i * CD;
        const float* ng = ode_ng + (size_t)i * CD;
        const float* nb = ode_nb + (size_t)i * CD;

        // stage this depth's ODE weights as f16 once (each reused 40x below)
        f32_to_f16_kernel<<<EWG, 256, 0, stream>>>(ode_w1 + (size_t)i * CD * CD, wh1, CD * CD);
        f32_to_f16_kernel<<<EWG, 256, 0, stream>>>(ode_w2 + (size_t)i * CD * CD, wh2, CD * CD);

        hipMemcpyAsync(ybuf, xs, TOK_BYTES, hipMemcpyDeviceToDevice, stream);
        hipMemcpyAsync(oout, xs, TOK_BYTES, hipMemcpyDeviceToDevice, stream);

        auto feval = [&](const float* yin, float* kout) {
            launch_gemm_wf16(yin, wh1, b1, t1, CM, CD, CD, 1, stream);      // relu fused
            launch_gemm_wf16(t1, wh2, b2, t2, CM, CD, CD, 0, stream);
            ln_kernel<<<CM, 256, 0, stream>>>(kout, t2, yin, ng, nb);       // LN(t2 + y)
        };
        for (int s = 0; s < 10; ++s) {
            feval(ybuf, k1);
            axpy_kernel<<<EWG, 256, 0, stream>>>(ytmp, ybuf, k1, 0.5f * CRES, NTOK);
            feval(ytmp, k2);
            axpy_kernel<<<EWG, 256, 0, stream>>>(ytmp, ybuf, k2, 0.5f * CRES, NTOK);
            feval(ytmp, k3);
            axpy_kernel<<<EWG, 256, 0, stream>>>(ytmp, ybuf, k3, CRES, NTOK);
            feval(ytmp, k4);
            rk4_kernel<<<EWG, 256, 0, stream>>>(ybuf, k1, k2, k3, k4, oout, lead, s);
        }
        ln_kernel<<<CM, 256, 0, stream>>>(xs, oout, xs, n2_g + (size_t)i * CD, n2_b + (size_t)i * CD);
    }

    // ---- phase 3: final norm, MLP head, unpatchify ----
    ln_kernel<<<CM, 256, 0, stream>>>(xs, xs, nullptr, norm_g, norm_b);
    launch_gemm(xs, h1_w, h1_b, hb1, CM, CD, CD, 2, stream);   // exact gelu
    launch_gemm(hb1, h2_w, h2_b, hb2, CM, CD, CD, 2, stream);  // exact gelu
    head_kernel<<<(CM * 20 + 255) / 256, 256, 0, stream>>>(hb2, hf_w, hf_b, outp);
}
